// PropagationGraphEncoder_34153579938344
// MI455X (gfx1250) — matrix operand order repaired and checked by reading, not compiled, not run on hardware
//
#include <hip/hip_runtime.h>
#include <hip/hip_bf16.h>

// ---------------------------------------------------------------------------
// PropagationGraphEncoder on MI455X (gfx1250, wave32, WMMA)
//   GEMMs       -> v_wmma_f32_16x16x32_bf16, double-buffered LDS, b128 staging
//   edge softmax-> global_atomic_max_u32 on a monotone float key
//   scatter     -> global_atomic_add_f32 (bandwidth-bound, ~2.2 GB traffic)
//   node dots   -> ds_add_f32 LDS atomics
// ---------------------------------------------------------------------------

typedef __attribute__((ext_vector_type(16))) __bf16          v16bf;
typedef __attribute__((ext_vector_type(8)))  float           v8f;
typedef __attribute__((ext_vector_type(16))) unsigned short  us16;

#define LN_EPS 1e-5f

// ---------------- helpers ----------------
__device__ __forceinline__ unsigned short f2bf(float f) {
  // round-to-nearest-even fp32 -> bf16
  unsigned int u = __float_as_uint(f);
  u += 0x7FFFu + ((u >> 16) & 1u);
  return (unsigned short)(u >> 16);
}

// monotone float -> uint key (order-preserving), for atomic max on floats
__device__ __forceinline__ unsigned int fkey(float f) {
  unsigned int i = __float_as_uint(f);
  return (i & 0x80000000u) ? ~i : (i | 0x80000000u);
}
__device__ __forceinline__ float fkey_dec(unsigned int k) {
  return (k & 0x80000000u) ? __uint_as_float(k & 0x7FFFFFFFu)
                           : __uint_as_float(~k);
}

__device__ __forceinline__ float elu(float v) {
  return v > 0.f ? v : (__expf(v) - 1.f);
}

// ---------------- elementwise utility kernels ----------------
__global__ void k_f32_to_bf16(const float* __restrict__ in,
                              unsigned short* __restrict__ out, int n) {
  int i = blockIdx.x * blockDim.x + threadIdx.x;
  if (i < n) out[i] = f2bf(in[i]);
}

__global__ void k_zero_u32(unsigned int* __restrict__ p, int n) {
  int i = blockIdx.x * blockDim.x + threadIdx.x;
  if (i < n) p[i] = 0u;
}

// ---------------- WMMA bf16 GEMM:  C(MxN) = A(MxK) * Wt(NxK)^T ----------------
// Block tile 128x128, BK=32, 256 threads = 8 waves, wave tile 32x64 (2x4 WMMA).
// Double-buffered LDS; b128 global staging with row-clamp (no exec branches).
// A, Wt are bf16 (ushort) row-major; C is f32 row-major.
// Requires: N % 128 == 0, K % 32 == 0 (true for 1024/768, 256/1024, 768/128).
__device__ __forceinline__ v16bf frag_a(const unsigned short* __restrict__ t,
                                        int rowbase, int lane) {
  // ISA 7.12.2, 16-bit A 16x32: lanes 0-15 K=0..7 (v0..3) & 16..23 (v4..7),
  // lanes 16-31 same +8.
  const int half = lane >> 4;
  const unsigned short* r = t + (rowbase + (lane & 15)) * 34;
  us16 u;
#pragma unroll
  for (int v = 0; v < 8; ++v) {
    int k0 = (v < 4 ? 2 * v : 8 + 2 * v) + half * 8;
    unsigned int pair = *(const unsigned int*)(r + k0);
    u[2 * v]     = (unsigned short)(pair & 0xFFFFu);
    u[2 * v + 1] = (unsigned short)(pair >> 16);
  }
  return __builtin_bit_cast(v16bf, u);
}

__device__ __forceinline__ v16bf frag_b(const unsigned short* __restrict__ t,
                                        int rowbase, int lane) {
  // 16-bit B 32x16 (staged transposed as [n][k]): lanes 0-15 K=0..7 & 16..23,
  // lanes 16-31 same +8 (matches frag_a).
  const int half = lane >> 4;
  const unsigned short* r = t + (rowbase + (lane & 15)) * 34;
  us16 u;
#pragma unroll
  for (int v = 0; v < 8; ++v) {
    int k0 = (v < 4 ? 2 * v : 8 + 2 * v) + half * 8;
    unsigned int pair = *(const unsigned int*)(r + k0);
    u[2 * v]     = (unsigned short)(pair & 0xFFFFu);
    u[2 * v + 1] = (unsigned short)(pair >> 16);
  }
  return __builtin_bit_cast(v16bf, u);
}

__global__ __launch_bounds__(256) void k_wmma_gemm(
    const unsigned short* __restrict__ A, const unsigned short* __restrict__ Wt,
    float* __restrict__ C, int M, int N, int K) {
  constexpr int BM = 128, BN = 128, BK = 32, ST = 34;
  __shared__ unsigned short As[2][BM * ST];
  __shared__ unsigned short Bs[2][BN * ST];

  const int tid  = threadIdx.x;
  const int lane = tid & 31;
  const int wid  = tid >> 5;
  const int wm   = wid >> 1;   // 0..3 -> 32-row strip
  const int wn   = wid & 1;    // 0..1 -> 64-col strip
  const int m0   = blockIdx.y * BM;
  const int n0   = blockIdx.x * BN;

  // staging coords: each thread covers rows r0 and r0+64, one uint4 (8 bf16) each
  const int r0 = tid >> 2;            // 0..63
  const int c8 = (tid & 3) * 8;       // ushort col within 32-wide K tile
  const int ra0 = min(m0 + r0, M - 1);       // clamp instead of predicate:
  const int ra1 = min(m0 + r0 + 64, M - 1);  // OOB rows never stored in epilogue
  const int rb0 = n0 + r0;                   // N % 128 == 0 -> always in range
  const int rb1 = n0 + r0 + 64;

  v8f acc[2][4];
  const v8f vz = {0.f, 0.f, 0.f, 0.f, 0.f, 0.f, 0.f, 0.f};
#pragma unroll
  for (int i = 0; i < 2; ++i)
#pragma unroll
    for (int j = 0; j < 4; ++j) acc[i][j] = vz;

  auto fetch = [&](int k0, uint4& a0, uint4& a1, uint4& b0, uint4& b1) {
    a0 = *(const uint4*)(A  + (size_t)ra0 * K + k0 + c8);
    a1 = *(const uint4*)(A  + (size_t)ra1 * K + k0 + c8);
    b0 = *(const uint4*)(Wt + (size_t)rb0 * K + k0 + c8);
    b1 = *(const uint4*)(Wt + (size_t)rb1 * K + k0 + c8);
  };
  auto stash = [&](int buf, const uint4& a0, const uint4& a1,
                   const uint4& b0, const uint4& b1) {
    unsigned int* p;
    p = (unsigned int*)(As[buf] + r0 * ST + c8);
    p[0] = a0.x; p[1] = a0.y; p[2] = a0.z; p[3] = a0.w;
    p = (unsigned int*)(As[buf] + (r0 + 64) * ST + c8);
    p[0] = a1.x; p[1] = a1.y; p[2] = a1.z; p[3] = a1.w;
    p = (unsigned int*)(Bs[buf] + r0 * ST + c8);
    p[0] = b0.x; p[1] = b0.y; p[2] = b0.z; p[3] = b0.w;
    p = (unsigned int*)(Bs[buf] + (r0 + 64) * ST + c8);
    p[0] = b1.x; p[1] = b1.y; p[2] = b1.z; p[3] = b1.w;
  };

  // prologue: tile 0 -> buffer 0
  uint4 ca0, ca1, cb0, cb1;
  fetch(0, ca0, ca1, cb0, cb1);
  stash(0, ca0, ca1, cb0, cb1);
  int buf = 0;

  for (int k0 = 0; k0 < K; k0 += BK) {
    __syncthreads();
    // start global loads for the next tile while WMMAs run on this one
    const bool more = (k0 + BK) < K;
    uint4 na0, na1, nb0, nb1;
    if (more) fetch(k0 + BK, na0, na1, nb0, nb1);
    if (k0 + 2 * BK < K) {   // speculative prefetch of tile t+2 (global_prefetch_b8)
      __builtin_prefetch(A  + (size_t)ra0 * K + k0 + 2 * BK, 0, 1);
      __builtin_prefetch(Wt + (size_t)rb0 * K + k0 + 2 * BK, 0, 1);
    }

    v16bf af[2], bfg[4];
#pragma unroll
    for (int tm = 0; tm < 2; ++tm) af[tm] = frag_a(As[buf], wm * 32 + tm * 16, lane);
#pragma unroll
    for (int tn = 0; tn < 4; ++tn) bfg[tn] = frag_b(Bs[buf], wn * 64 + tn * 16, lane);

#pragma unroll
    for (int tm = 0; tm < 2; ++tm)
#pragma unroll
      for (int tn = 0; tn < 4; ++tn)
        acc[tm][tn] = __builtin_amdgcn_wmma_f32_16x16x32_bf16(
            false, af[tm], false, bfg[tn], (short)0, acc[tm][tn], false, false);

    if (more) stash(buf ^ 1, na0, na1, nb0, nb1);
    buf ^= 1;
  }

  // epilogue: C/D layout -> VGPR r holds (M=r, lanes 0-15) and (M=8+r, lanes 16-31)
  const int half = lane >> 4;
  const int cl   = lane & 15;
#pragma unroll
  for (int tm = 0; tm < 2; ++tm)
#pragma unroll
    for (int tn = 0; tn < 4; ++tn)
#pragma unroll
      for (int r = 0; r < 8; ++r) {
        int row = m0 + wm * 32 + tm * 16 + half * 8 + r;
        int col = n0 + wn * 64 + tn * 16 + cl;
        if (row < M) C[(size_t)row * N + col] = acc[tm][tn][r];
      }
}

// ---------------- per-node attention dot products ----------------
// alpha_s[n,h] = sum_o h[n,h,o]*a_src[h,o]  (flat channel index == flat a index)
__global__ void k_alpha(const float* __restrict__ h, const float* __restrict__ a_s,
                        const float* __restrict__ a_d, float* __restrict__ as,
                        float* __restrict__ ad, int H, int HD) {
  __shared__ float ss[4], sd[4];
  int n = blockIdx.x, t = threadIdx.x;
  if (t < H) { ss[t] = 0.f; sd[t] = 0.f; }
  __syncthreads();
  int OUTd = HD / H;
  for (int c = t; c < HD; c += blockDim.x) {
    float hv = h[(size_t)n * HD + c];
    int hd = c / OUTd;
    atomicAdd(&ss[hd], hv * a_s[c]);   // ds_add_f32
    atomicAdd(&sd[hd], hv * a_d[c]);
  }
  __syncthreads();
  if (t < H) { as[n * H + t] = ss[t]; ad[n * H + t] = sd[t]; }
}

// ---------------- edge softmax passes ----------------
// edges 0..E-1 from edge_index, edges E..E+N-1 are self-loops
__global__ void k_edge_max(const int* __restrict__ ei, const float* __restrict__ as,
                           const float* __restrict__ ad, float* __restrict__ ev,
                           unsigned int* __restrict__ mkey, int E, int Nn, int H) {
  int id = blockIdx.x * blockDim.x + threadIdx.x;
  int total = (E + Nn) * H;
  if (id >= total) return;
  int hd = id % H, e = id / H;
  int s, d;
  if (e < E) { s = ei[e]; d = ei[E + e]; } else { s = e - E; d = s; }
  float v = as[s * H + hd] + ad[d * H + hd];
  v = v > 0.f ? v : 0.2f * v;            // leaky_relu(0.2)
  ev[id] = v;
  atomicMax(&mkey[d * H + hd], fkey(v)); // global_atomic_max_u32
}

__global__ void k_edge_exp(const int* __restrict__ ei, float* __restrict__ ev,
                           const unsigned int* __restrict__ mkey,
                           float* __restrict__ den, int E, int Nn, int H) {
  int id = blockIdx.x * blockDim.x + threadIdx.x;
  int total = (E + Nn) * H;
  if (id >= total) return;
  int hd = id % H, e = id / H;
  int d;
  if (e < E) { d = ei[E + e]; } else { d = e - E; }
  float ex = __expf(ev[id] - fkey_dec(mkey[d * H + hd]));
  ev[id] = ex;
  atomicAdd(&den[d * H + hd], ex);
}

// one block per edge: gather h[src], scale by alpha, scatter-add into agg[dst]
__global__ void k_edge_scatter(const int* __restrict__ ei, const float* __restrict__ h,
                               const float* __restrict__ ev, const float* __restrict__ den,
                               float* __restrict__ agg, int E, int Nn, int H, int HD) {
  int e = blockIdx.x;
  int s, d;
  if (e < E) { s = ei[e]; d = ei[E + e]; } else { s = e - E; d = s; }
  int OUTd = HD / H;
  for (int c = threadIdx.x; c < HD; c += blockDim.x) {
    int hd = c / OUTd;
    float w = ev[(size_t)e * H + hd] / (den[d * H + hd] + 1e-16f);
    atomicAdd(&agg[(size_t)d * HD + c], h[(size_t)s * HD + c] * w);
  }
}

// ---------------- layer epilogues ----------------
// layer1: h1 = elu(agg + b1) -> bf16 (input to GEMM2)
__global__ void k_bias_elu_bf16(const float* __restrict__ agg, const float* __restrict__ b,
                                unsigned short* __restrict__ outb, int total, int HD) {
  int i = blockIdx.x * blockDim.x + threadIdx.x;
  if (i >= total) return;
  float v = agg[i] + b[i % HD];
  outb[i] = f2bf(elu(v));
}

// layer2: mean over heads + b2, elu -> h2 (f32, directly into d_out region)
__global__ void k_headmean_bias_elu(const float* __restrict__ agg, const float* __restrict__ b,
                                    float* __restrict__ h2, int Nn, int OUTd, int H) {
  int i = blockIdx.x * blockDim.x + threadIdx.x;
  if (i >= Nn * OUTd) return;
  int n = i / OUTd, o = i % OUTd;
  float s = 0.f;
  for (int hd = 0; hd < H; ++hd) s += agg[(size_t)n * H * OUTd + hd * OUTd + o];
  h2[i] = elu(s / (float)H + b[o]);
}

// ---------------- pooling ----------------
__global__ void k_pool(const float* __restrict__ h2, const int* __restrict__ batch,
                       float* __restrict__ pooled, unsigned int* __restrict__ cnt,
                       int Nn, int F) {
  int i = blockIdx.x * blockDim.x + threadIdx.x;
  if (i >= Nn * F) return;
  int n = i / F, o = i % F;
  int b = batch[n];
  atomicAdd(&pooled[b * F + o], h2[i]);
  if (o == 0) atomicAdd(&cnt[b], 1u);
}

__global__ void k_pool_fin(const float* __restrict__ pooled,
                           const unsigned int* __restrict__ cnt,
                           unsigned short* __restrict__ pb, int B, int F) {
  int i = blockIdx.x * blockDim.x + threadIdx.x;
  if (i >= B * F) return;
  int b = i / F;
  unsigned int c = cnt[b];
  float cf = (float)(c > 0u ? c : 1u);
  pb[i] = f2bf(pooled[i] / cf);
}

// ---------------- projection bias + LayerNorm ----------------
__global__ __launch_bounds__(256) void k_proj_ln(
    const float* __restrict__ z, const float* __restrict__ pb,
    const float* __restrict__ g, const float* __restrict__ beta,
    float* __restrict__ out) {
  __shared__ float zb[768];
  __shared__ float red[256];
  int row = blockIdx.x, t = threadIdx.x;
  float psum = 0.f;
  for (int j = t; j < 768; j += 256) {
    float v = z[row * 768 + j] + pb[j];
    zb[j] = v;
    psum += v;
  }
  red[t] = psum; __syncthreads();
  for (int s = 128; s > 0; s >>= 1) { if (t < s) red[t] += red[t + s]; __syncthreads(); }
  float mu = red[0] / 768.f;
  __syncthreads();
  float pv = 0.f;
  for (int j = t; j < 768; j += 256) { float d = zb[j] - mu; pv += d * d; }
  red[t] = pv; __syncthreads();
  for (int s = 128; s > 0; s >>= 1) { if (t < s) red[t] += red[t + s]; __syncthreads(); }
  float rstd = rsqrtf(red[0] / 768.f + LN_EPS);
  for (int j = t; j < 768; j += 256)
    out[row * 768 + j] = (zb[j] - mu) * rstd * g[j] + beta[j];
}

// ---------------------------------------------------------------------------
extern "C" void kernel_launch(void* const* d_in, const int* in_sizes, int n_in,
                              void* d_out, int out_size, void* d_ws, size_t ws_size,
                              hipStream_t stream) {
  const int N = 16000, E = 256000, B = 32;
  const int IN = 768, HID = 256, H1 = 4, OUTd = 128, H2 = 2;
  const int HD1 = H1 * HID;   // 1024
  const int HD2 = H2 * OUTd;  // 256
  const int EP = E + N;       // edges + self loops

  const float* x   = (const float*)d_in[0];
  const int*   ei  = (const int*)d_in[1];
  const int*   bat = (const int*)d_in[2];
  const float* W1  = (const float*)d_in[3];
  const float* a1s = (const float*)d_in[4];
  const float* a1d = (const float*)d_in[5];
  const float* b1  = (const float*)d_in[6];
  const float* W2  = (const float*)d_in[7];
  const float* a2s = (const float*)d_in[8];
  const float* a2d = (const float*)d_in[9];
  const float* b2  = (const float*)d_in[10];
  const float* pW  = (const float*)d_in[11];
  const float* pb  = (const float*)d_in[12];
  const float* lg  = (const float*)d_in[13];
  const float* lb  = (const float*)d_in[14];

  float* ge_out = (float*)d_out;              // (32,768)
  float* h2_out = (float*)d_out + B * 768;    // (16000,128)

  // ---- workspace bump allocator (256B aligned) ----
  char* wsb = (char*)d_ws;
  size_t off = 0;
  auto alloc = [&](size_t bytes) -> void* {
    void* p = wsb + off;
    off = (off + bytes + 255) & ~(size_t)255;
    return p;
  };
  unsigned short* xb   = (unsigned short*)alloc((size_t)N * IN * 2);
  unsigned short* W1b  = (unsigned short*)alloc((size_t)HD1 * IN * 2);
  float* hL1           = (float*)alloc((size_t)N * HD1 * 4);
  float* as1           = (float*)alloc((size_t)N * H1 * 4);
  float* ad1           = (float*)alloc((size_t)N * H1 * 4);
  unsigned int* m1     = (unsigned int*)alloc((size_t)N * H1 * 4);
  float* den1          = (float*)alloc((size_t)N * H1 * 4);
  float* ev1           = (float*)alloc((size_t)EP * H1 * 4);
  float* agg1          = (float*)alloc((size_t)N * HD1 * 4);
  unsigned short* h1b  = (unsigned short*)alloc((size_t)N * HD1 * 2);
  unsigned short* W2b  = (unsigned short*)alloc((size_t)HD2 * HD1 * 2);
  float* hL2           = (float*)alloc((size_t)N * HD2 * 4);
  float* as2           = (float*)alloc((size_t)N * H2 * 4);
  float* ad2           = (float*)alloc((size_t)N * H2 * 4);
  unsigned int* m2     = (unsigned int*)alloc((size_t)N * H2 * 4);
  float* den2          = (float*)alloc((size_t)N * H2 * 4);
  float* ev2           = (float*)alloc((size_t)EP * H2 * 4);
  float* agg2          = (float*)alloc((size_t)N * HD2 * 4);
  float* pooled        = (float*)alloc((size_t)B * OUTd * 4);
  unsigned int* cnt    = (unsigned int*)alloc((size_t)B * 4);
  unsigned short* pdb  = (unsigned short*)alloc((size_t)B * OUTd * 2);
  unsigned short* pWb  = (unsigned short*)alloc((size_t)768 * OUTd * 2);
  float* z             = (float*)alloc((size_t)B * 768 * 4);

  auto cdiv = [](int a, int b) { return (a + b - 1) / b; };
  const dim3 T(256);

  // ---- convert GEMM operands to bf16 ----
  k_f32_to_bf16<<<cdiv(N * IN, 256), T, 0, stream>>>(x, xb, N * IN);
  k_f32_to_bf16<<<cdiv(HD1 * IN, 256), T, 0, stream>>>(W1, W1b, HD1 * IN);
  k_f32_to_bf16<<<cdiv(HD2 * HD1, 256), T, 0, stream>>>(W2, W2b, HD2 * HD1);
  k_f32_to_bf16<<<cdiv(768 * OUTd, 256), T, 0, stream>>>(pW, pWb, 768 * OUTd);

  // ---- GEMM1: hL1 = x @ W1^T  (16000x768 * 768x1024) ----
  k_wmma_gemm<<<dim3(HD1 / 128, cdiv(N, 128)), T, 0, stream>>>(xb, W1b, hL1, N, HD1, IN);

  // ---- GAT layer 1 ----
  k_alpha<<<N, dim3(128), 0, stream>>>(hL1, a1s, a1d, as1, ad1, H1, HD1);
  k_zero_u32<<<cdiv(N * H1, 256), T, 0, stream>>>(m1, N * H1);
  k_zero_u32<<<cdiv(N * H1, 256), T, 0, stream>>>((unsigned int*)den1, N * H1);
  k_zero_u32<<<cdiv(N * HD1, 256), T, 0, stream>>>((unsigned int*)agg1, N * HD1);
  k_edge_max<<<cdiv(EP * H1, 256), T, 0, stream>>>(ei, as1, ad1, ev1, m1, E, N, H1);
  k_edge_exp<<<cdiv(EP * H1, 256), T, 0, stream>>>(ei, ev1, m1, den1, E, N, H1);
  k_edge_scatter<<<EP, T, 0, stream>>>(ei, hL1, ev1, den1, agg1, E, N, H1, HD1);
  k_bias_elu_bf16<<<cdiv(N * HD1, 256), T, 0, stream>>>(agg1, b1, h1b, N * HD1, HD1);

  // ---- GEMM2: hL2 = h1 @ W2^T  (16000x1024 * 1024x256) ----
  k_wmma_gemm<<<dim3(HD2 / 128, cdiv(N, 128)), T, 0, stream>>>(h1b, W2b, hL2, N, HD2, HD1);

  // ---- GAT layer 2 ----
  k_alpha<<<N, dim3(128), 0, stream>>>(hL2, a2s, a2d, as2, ad2, H2, HD2);
  k_zero_u32<<<cdiv(N * H2, 256), T, 0, stream>>>(m2, N * H2);
  k_zero_u32<<<cdiv(N * H2, 256), T, 0, stream>>>((unsigned int*)den2, N * H2);
  k_zero_u32<<<cdiv(N * HD2, 256), T, 0, stream>>>((unsigned int*)agg2, N * HD2);
  k_edge_max<<<cdiv(EP * H2, 256), T, 0, stream>>>(ei, as2, ad2, ev2, m2, E, N, H2);
  k_edge_exp<<<cdiv(EP * H2, 256), T, 0, stream>>>(ei, ev2, m2, den2, E, N, H2);
  k_edge_scatter<<<EP, T, 0, stream>>>(ei, hL2, ev2, den2, agg2, E, N, H2, HD2);
  k_headmean_bias_elu<<<cdiv(N * OUTd, 256), T, 0, stream>>>(agg2, b2, h2_out, N, OUTd, H2);

  // ---- global mean pool ----
  k_zero_u32<<<cdiv(B * OUTd, 256), T, 0, stream>>>((unsigned int*)pooled, B * OUTd);
  k_zero_u32<<<1, T, 0, stream>>>(cnt, B);
  k_pool<<<cdiv(N * OUTd, 256), T, 0, stream>>>(h2_out, bat, pooled, cnt, N, OUTd);
  k_pool_fin<<<cdiv(B * OUTd, 256), T, 0, stream>>>(pooled, cnt, pdb, B, OUTd);

  // ---- GEMM3: z = pooled @ proj_W^T  (32x128 * 128x768) ----
  k_wmma_gemm<<<dim3(768 / 128, cdiv(B, 128)), T, 0, stream>>>(pdb, pWb, z, B, 768, OUTd);

  // ---- bias + LayerNorm -> graph_embed ----
  k_proj_ln<<<B, T, 0, stream>>>(z, pb, lg, lb, ge_out);
  (void)in_sizes; (void)n_in; (void)out_size; (void)ws_size;
}
